// MultiHeadAttention_79912161509819
// MI455X (gfx1250) — matrix operand order repaired and checked by reading, not compiled, not run on hardware
//
#include <hip/hip_runtime.h>
#include <hip/hip_bf16.h>

// MHA forward for MI455X (gfx1250, wave32, WMMA).
// Pipeline: cvt(x,W) -> WMMA GEMM qkv -> flash-attention (WMMA) -> WMMA GEMM out.

typedef _Float16 f16;
typedef _Float16 v8h  __attribute__((ext_vector_type(8)));
typedef _Float16 v16h __attribute__((ext_vector_type(16)));
typedef float    v8f  __attribute__((ext_vector_type(8)));

#define D_MODEL 768
#define N_HEADS 12
#define D_HEAD  64
#define SEQ     2048
#define BATCH   2
#define MTOT    (BATCH*SEQ)   // 4096
#define NQKV    (3*D_MODEL)   // 2304

#define USE_ASYNC_LDS 1

__device__ __forceinline__ v16h cat16(v8h lo, v8h hi) {
  return __builtin_shufflevector(lo, hi, 0,1,2,3,4,5,6,7,8,9,10,11,12,13,14,15);
}
__device__ __forceinline__ v8f wmma_f16(v16h a, v16h b, v8f c) {
  return __builtin_amdgcn_wmma_f32_16x16x32_f16(false, a, false, b, (short)0, c, false, false);
}

// hardware exp2 (v_exp_f32)
__device__ __forceinline__ float fast_exp2(float x) {
#if __has_builtin(__builtin_amdgcn_exp2f)
  return __builtin_amdgcn_exp2f(x);
#else
  return exp2f(x);
#endif
}

// lanes 0-15 <-> 16-31 swap, pure VALU (v_permlanex16_b32, identity selects)
__device__ __forceinline__ float swap16(float x) {
#if __has_builtin(__builtin_amdgcn_permlanex16)
  unsigned xi = (unsigned)__float_as_int(x);
  unsigned r = __builtin_amdgcn_permlanex16(xi, xi, 0x76543210u, 0xfedcba98u, false, false);
  return __int_as_float((int)r);
#else
  return __shfl_xor(x, 16, 32);
#endif
}

#if USE_ASYNC_LDS
// 16B/lane direct global->LDS copy, tracked by ASYNCcnt (CDNA5)
__device__ __forceinline__ void async_b128(void* lds, const void* g) {
  unsigned loff = (unsigned)(uintptr_t)lds;              // LDS aperture: addr[31:0]
  unsigned long long ga = (unsigned long long)(uintptr_t)g;
  asm volatile("global_load_async_to_lds_b128 %0, %1, off"
               :: "v"(loff), "v"(ga) : "memory");
}
__device__ __forceinline__ void wait_async0() {
  asm volatile("s_wait_asynccnt 0x0" ::: "memory");
}
#else
__device__ __forceinline__ void wait_async0() {}
#endif

// stage one 16-f16 strip of A and B into LDS (per thread)
__device__ __forceinline__ void stage_tile(f16* lA, f16* lB, const f16* gA, const f16* gB) {
#if USE_ASYNC_LDS
  async_b128(lA,     gA);
  async_b128(lA + 8, gA + 8);
  async_b128(lB,     gB);
  async_b128(lB + 8, gB + 8);
#else
  *(v8h*)lA       = *(const v8h*)gA;
  *(v8h*)(lA + 8) = *(const v8h*)(gA + 8);
  *(v8h*)lB       = *(const v8h*)gB;
  *(v8h*)(lB + 8) = *(const v8h*)(gB + 8);
#endif
}

// ---------------- conversion kernels ----------------
__global__ void k_cvt(const float* __restrict__ in, f16* __restrict__ out, int n) {
  int i = blockIdx.x * blockDim.x + threadIdx.x;
  if (i < n) out[i] = (f16)in[i];
}
// in: [K,N] f32 row-major -> out: [N,K] f16 row-major
__global__ void k_cvt_t(const float* __restrict__ in, f16* __restrict__ out, int K, int N) {
  int i = blockIdx.x * blockDim.x + threadIdx.x;
  if (i < K * N) { int k = i / N, n = i % N; out[n * K + k] = (f16)in[i]; }
}

// ---------------- WMMA GEMM: C[M,N] = A[M,K] @ BT[N,K]^T + bias ----------------
// Double-buffered async-to-LDS pipeline: DMA of tile k+1 overlaps WMMAs of tile k.
// MODE 0: scatter f16 into per-head Q[p,dh], K[p,dh], VT[dh,p]
// MODE 1: fp32 output (final projection)
template<int MODE>
__global__ __launch_bounds__(256)
void k_gemm(const f16* __restrict__ A, const f16* __restrict__ BT,
            const float* __restrict__ bias, int K, int N,
            float* __restrict__ outF,
            f16* __restrict__ qO, f16* __restrict__ kO, f16* __restrict__ vTO)
{
  __shared__ f16 sA[2][128 * 40];   // 128 rows x 32 K, stride 40 (b128-aligned pad)
  __shared__ f16 sB[2][128 * 40];

  const int tid  = threadIdx.x;
  const int lane = tid & 31;
  const int w    = tid >> 5;     // wave 0..7
  const int wm   = w >> 2;       // 0..1  (64 rows each)
  const int wn   = w & 3;        // 0..3  (32 cols each)
  const int bm   = blockIdx.y * 128;
  const int bn   = blockIdx.x * 128;
  const int l15  = lane & 15;
  const int hi16 = lane >> 4;
  const int kbA  = hi16 * 16;    // A-fragment K base (contiguous 16)
  const int kbB  = hi16 * 16;    // B-fragment K base (contiguous 16)

  v8f acc[4][2];
  #pragma unroll
  for (int i = 0; i < 4; i++)
    #pragma unroll
    for (int j = 0; j < 2; j++)
      #pragma unroll
      for (int g = 0; g < 8; g++) acc[i][j][g] = 0.0f;

  const int lr = tid >> 1;            // 0..127 LDS row
  const int lh = (tid & 1) * 16;      // 0 / 16 column group
  const f16* gA = A  + (size_t)(bm + lr) * K + lh;
  const f16* gB = BT + (size_t)(bn + lr) * K + lh;
  f16* lAw[2] = { &sA[0][lr * 40 + lh], &sA[1][lr * 40 + lh] };
  f16* lBw[2] = { &sB[0][lr * 40 + lh], &sB[1][lr * 40 + lh] };

  // prologue: stage tile 0
  stage_tile(lAw[0], lBw[0], gA, gB);
  wait_async0();
  __syncthreads();

  int buf = 0;
  for (int k0 = 0; k0 < K; k0 += 32) {
    // kick off DMA for next tile into the other buffer (overlaps compute)
    if (k0 + 32 < K)
      stage_tile(lAw[buf ^ 1], lBw[buf ^ 1], gA + k0 + 32, gB + k0 + 32);

    const f16* cA = sA[buf];
    const f16* cB = sB[buf];
    v16h bf[2];
    #pragma unroll
    for (int j = 0; j < 2; j++) {
      int n = wn * 32 + j * 16 + l15;
      bf[j] = cat16(*(const v8h*)&cB[n * 40 + kbB],
                    *(const v8h*)&cB[n * 40 + kbB + 8]);
    }
    #pragma unroll
    for (int i = 0; i < 4; i++) {
      int m = wm * 64 + i * 16 + l15;
      v16h af = cat16(*(const v8h*)&cA[m * 40 + kbA],
                      *(const v8h*)&cA[m * 40 + kbA + 8]);
      #pragma unroll
      for (int j = 0; j < 2; j++)
        acc[i][j] = wmma_f16(af, bf[j], acc[i][j]);
    }

    wait_async0();        // next tile fully in LDS (own wave)...
    __syncthreads();      // ...and for all waves
    buf ^= 1;
  }

  // epilogue
  const int mh = hi16 * 8;
  #pragma unroll
  for (int i = 0; i < 4; i++) {
    #pragma unroll
    for (int j = 0; j < 2; j++) {
      int n = bn + wn * 32 + j * 16 + l15;
      float bv = bias[n];
      int mbase = bm + wm * 64 + i * 16 + mh;
      if constexpr (MODE == 1) {
        #pragma unroll
        for (int g = 0; g < 8; g++)
          outF[(size_t)(mbase + g) * N + n] = acc[i][j][g] + bv;
      } else {
        int h = n / (3 * D_HEAD), wc = n % (3 * D_HEAD);
        #pragma unroll
        for (int g = 0; g < 8; g++) {
          int m  = mbase + g;
          int bb = m >> 11, tok = m & (SEQ - 1);
          f16 val = (f16)(acc[i][j][g] + bv);
          size_t ho = (size_t)(bb * N_HEADS + h);
          if (wc < D_HEAD)           qO [(ho * SEQ + tok) * D_HEAD + wc]             = val;
          else if (wc < 2 * D_HEAD)  kO [(ho * SEQ + tok) * D_HEAD + (wc - D_HEAD)]  = val;
          else                       vTO[(ho * D_HEAD + (wc - 2 * D_HEAD)) * SEQ + tok] = val;
        }
      }
    }
  }
}

// ---------------- flash attention (per wave: 16 queries x all keys) ----------------
// S^T = K @ Q^T (keys=M, queries=N): softmax-over-keys is per-lane + one lane-swap.
// Q pre-scaled by 8*log2(e) (faithful source bug: logits *= sqrt(d_head)) so the
// softmax runs in the log2 domain and exp is a bare v_exp_f32.
__global__ __launch_bounds__(256)
void k_attn(const f16* __restrict__ Q, const f16* __restrict__ Kt,
            const f16* __restrict__ Vt, f16* __restrict__ O)
{
  const int tid  = threadIdx.x;
  const int lane = tid & 31;
  const int w    = tid >> 5;
  const int bh   = blockIdx.y;
  const int bb   = bh / N_HEADS, h = bh % N_HEADS;
  const int q0   = blockIdx.x * 128 + w * 16;
  const int l15  = lane & 15;
  const int hi16 = lane >> 4;

  const f16* Qp = Q  + (size_t)bh * SEQ * D_HEAD;
  const f16* Kp = Kt + (size_t)bh * SEQ * D_HEAD;
  const f16* Vp = Vt + (size_t)bh * D_HEAD * SEQ;

  // Q^T B-fragments (persistent), pre-scaled by 8*log2(e)
  v16h qf[2];
  {
    const f16* qrow = Qp + (size_t)(q0 + l15) * D_HEAD + hi16 * 16;
    const f16 qs = (f16)(8.0f * 1.4426950408889634f);
    qf[0] = cat16(*(const v8h*)(qrow + 0),  *(const v8h*)(qrow + 8))  * qs;
    qf[1] = cat16(*(const v8h*)(qrow + 32), *(const v8h*)(qrow + 40)) * qs;
  }

  v8f ot[4];
  #pragma unroll
  for (int t = 0; t < 4; t++)
    #pragma unroll
    for (int g = 0; g < 8; g++) ot[t][g] = 0.0f;
  float mrun = -3.0e38f, lrun = 0.0f;

  for (int kc = 0; kc < SEQ; kc += 64) {     // 64-key chunks: 4 S-tiles
    v8f s[4];
    #pragma unroll
    for (int st = 0; st < 4; st++) {
      const f16* krow = Kp + (size_t)(kc + st * 16 + l15) * D_HEAD + hi16 * 16;
      v16h kf0 = cat16(*(const v8h*)(krow + 0),  *(const v8h*)(krow + 8));
      v16h kf1 = cat16(*(const v8h*)(krow + 32), *(const v8h*)(krow + 40));
      v8f ss;
      #pragma unroll
      for (int g = 0; g < 8; g++) ss[g] = 0.0f;
      ss = wmma_f16(kf0, qf[0], ss);
      s[st] = wmma_f16(kf1, qf[1], ss);
    }

    // chunk max: 4 independent per-tile partials, then log-depth combine
    float pm[4];
    #pragma unroll
    for (int st = 0; st < 4; st++) {
      float m0 = fmaxf(fmaxf(s[st][0], s[st][1]), fmaxf(s[st][2], s[st][3]));
      float m1 = fmaxf(fmaxf(s[st][4], s[st][5]), fmaxf(s[st][6], s[st][7]));
      pm[st] = fmaxf(m0, m1);
    }
    float cm = fmaxf(fmaxf(pm[0], pm[1]), fmaxf(pm[2], pm[3]));
    cm = fmaxf(cm, swap16(cm));
    float mnew = fmaxf(mrun, cm);
    float corr = fast_exp2(mrun - mnew);

    // exp + sum (4 independent partial sums, log-depth combine)
    float prs[4];
    #pragma unroll
    for (int st = 0; st < 4; st++) {
      float e0 = fast_exp2(s[st][0] - mnew), e1 = fast_exp2(s[st][1] - mnew);
      float e2 = fast_exp2(s[st][2] - mnew), e3 = fast_exp2(s[st][3] - mnew);
      float e4 = fast_exp2(s[st][4] - mnew), e5 = fast_exp2(s[st][5] - mnew);
      float e6 = fast_exp2(s[st][6] - mnew), e7 = fast_exp2(s[st][7] - mnew);
      s[st][0] = e0; s[st][1] = e1; s[st][2] = e2; s[st][3] = e3;
      s[st][4] = e4; s[st][5] = e5; s[st][6] = e6; s[st][7] = e7;
      prs[st] = ((e0 + e1) + (e2 + e3)) + ((e4 + e5) + (e6 + e7));
    }
    float rs = (prs[0] + prs[1]) + (prs[2] + prs[3]);
    rs += swap16(rs);
    lrun = lrun * corr + rs;
    mrun = mnew;
    #pragma unroll
    for (int t = 0; t < 4; t++) ot[t] = ot[t] * corr;

    // build two P B-fragments (each 32 keys x 16 queries) from C-layout tiles
    v16h pf[2];
    bool lo = (lane < 16);
    #pragma unroll
    for (int half = 0; half < 2; half++) {
      v8f& sa = s[2 * half];       // keys +0..15 of this 32-group
      v8f& sb = s[2 * half + 1];   // keys +16..31
      #pragma unroll
      for (int i = 0; i < 4; i++) {             // frag K = 2i,2i+1 (rows 0..7)
        float ra = swap16(sb[2 * i]);
        float rb = swap16(sb[2 * i + 1]);
        pf[half][2 * i]     = (f16)(lo ? sa[2 * i]     : ra);
        pf[half][2 * i + 1] = (f16)(lo ? sa[2 * i + 1] : rb);
      }
      #pragma unroll
      for (int i = 4; i < 8; i++) {             // rows 8..15 live in partner lane
        float ra = swap16(sa[2 * i - 8]);
        float rb = swap16(sa[2 * i - 7]);
        pf[half][2 * i]     = (f16)(lo ? ra : sb[2 * i - 8]);
        pf[half][2 * i + 1] = (f16)(lo ? rb : sb[2 * i - 7]);
      }
    }

    // O^T += V^T @ P   (4 dh-tiles x 2 key-groups)
    #pragma unroll
    for (int t = 0; t < 4; t++) {
      const f16* vrow = Vp + (size_t)(16 * t + l15) * SEQ + kc + hi16 * 16;
      v16h vf0 = cat16(*(const v8h*)(vrow + 0),  *(const v8h*)(vrow + 8));
      v16h vf1 = cat16(*(const v8h*)(vrow + 32), *(const v8h*)(vrow + 40));
      ot[t] = wmma_f16(vf0, pf[0], ot[t]);
      ot[t] = wmma_f16(vf1, pf[1], ot[t]);
    }
  }

  // normalize and store O (f16, [token, 768])
  float inv = 1.0f / lrun;
  int token = bb * SEQ + q0 + l15;
  #pragma unroll
  for (int t = 0; t < 4; t++) {
    v8h pk;
    #pragma unroll
    for (int g = 0; g < 8; g++) pk[g] = (f16)(ot[t][g] * inv);
    int dhs = 16 * t + 8 * hi16;
    *(v8h*)(O + (size_t)token * D_MODEL + h * D_HEAD + dhs) = pk;
  }
}

// ---------------- launch ----------------
extern "C" void kernel_launch(void* const* d_in, const int* in_sizes, int n_in,
                              void* d_out, int out_size, void* d_ws, size_t ws_size,
                              hipStream_t stream)
{
  (void)in_sizes; (void)n_in; (void)out_size; (void)ws_size;
  const float* x    = (const float*)d_in[0];
  const float* Wqkv = (const float*)d_in[1];
  const float* bqkv = (const float*)d_in[2];
  const float* Wout = (const float*)d_in[3];
  const float* bout = (const float*)d_in[4];
  float* out = (float*)d_out;

  char* ws = (char*)d_ws;
  size_t off = 0;
  auto alloc = [&](size_t bytes) -> void* {
    void* p = ws + off;
    off = (off + bytes + 255) & ~(size_t)255;
    return p;
  };
  f16* xh     = (f16*)alloc((size_t)MTOT * D_MODEL * 2);
  f16* wqkvT  = (f16*)alloc((size_t)NQKV * D_MODEL * 2);
  f16* woutT  = (f16*)alloc((size_t)D_MODEL * D_MODEL * 2);
  f16* qB     = (f16*)alloc((size_t)BATCH * N_HEADS * SEQ * D_HEAD * 2);
  f16* kB     = (f16*)alloc((size_t)BATCH * N_HEADS * SEQ * D_HEAD * 2);
  f16* vTB    = (f16*)alloc((size_t)BATCH * N_HEADS * SEQ * D_HEAD * 2);
  f16* Oh     = (f16*)alloc((size_t)MTOT * D_MODEL * 2);

  // fp32 -> f16 conversions (+ weight transposes for contiguous-K B fragments)
  {
    int n = MTOT * D_MODEL;
    k_cvt<<<(n + 255) / 256, 256, 0, stream>>>(x, xh, n);
  }
  k_cvt_t<<<(D_MODEL * NQKV + 255) / 256, 256, 0, stream>>>(Wqkv, wqkvT, D_MODEL, NQKV);
  k_cvt_t<<<(D_MODEL * D_MODEL + 255) / 256, 256, 0, stream>>>(Wout, woutT, D_MODEL, D_MODEL);

  // QKV projection -> per-head Q, K, V^T (f16)
  k_gemm<0><<<dim3(NQKV / 128, MTOT / 128), 256, 0, stream>>>(
      xh, wqkvT, bqkv, D_MODEL, NQKV, nullptr, qB, kB, vTB);

  // flash attention
  k_attn<<<dim3(SEQ / 128, BATCH * N_HEADS), 256, 0, stream>>>(qB, kB, vTB, Oh);

  // output projection (fp32 out + bias)
  k_gemm<1><<<dim3(D_MODEL / 128, MTOT / 128), 256, 0, stream>>>(
      Oh, woutT, bout, D_MODEL, D_MODEL, out, nullptr, nullptr, nullptr);
}
